// AgentAwareAttentionV2_15264313770586
// MI455X (gfx1250) — matrix operand order repaired and checked by reading, not compiled, not run on hardware
//
#include <hip/hip_runtime.h>

// ---------------------------------------------------------------------------
// AgentAwareAttentionV2 for MI455X (gfx1250, wave32, WMMA bf16 16x16x32).
// Pipeline: 5x projection GEMMs (fp32->bf16, head-split layout) ->
//           flash attention w/ agent-aware score select + online softmax
//           (K/mask tiles staged via GLOBAL_LOAD_ASYNC_TO_LDS_B128) ->
//           second pass for head-averaged attention output (atomics) ->
//           final FC GEMM with bias.
// ---------------------------------------------------------------------------

#define DEV __device__ __forceinline__

typedef __bf16 bf16_t;
typedef bf16_t v16bf __attribute__((ext_vector_type(16)));
typedef float  v8f   __attribute__((ext_vector_type(8)));

union FragBF {
  unsigned int   u32[8];
  unsigned short u16[16];
  v16bf          v;
};

DEV unsigned short f2bf(float x) {
  unsigned u = __builtin_bit_cast(unsigned, x);
  u += 0x7FFFu + ((u >> 16) & 1u);   // round-to-nearest-even
  return (unsigned short)(u >> 16);
}

DEV v8f vzero() {
  v8f z = {0.f, 0.f, 0.f, 0.f, 0.f, 0.f, 0.f, 0.f};
  return z;
}

// A-matrix (16x32 bf16) per-lane gather from row-major source.
DEV void frag_load_a(FragBF& f, const unsigned int* row32, int half) {
#pragma unroll
  for (int g = 0; g < 4; ++g) {
    f.u32[g]     = row32[half * 4 + g];
    f.u32[4 + g] = row32[8 + half * 4 + g];
  }
}

// B-matrix (32x16 bf16) per-lane gather: lane holds column (lane&15),
// K order matches A (8h+s, then 16+8h+s).
DEV void frag_load_b(FragBF& f, const unsigned int* col32, int half) {
#pragma unroll
  for (int g = 0; g < 4; ++g) {
    f.u32[g]     = col32[half * 4 + g];
    f.u32[4 + g] = col32[8 + half * 4 + g];
  }
}

DEV v8f wmma_bf16(const FragBF& a, const FragBF& b, v8f c) {
  return __builtin_amdgcn_wmma_f32_16x16x32_bf16(
      /*neg_a=*/false, a.v, /*neg_b=*/false, b.v,
      /*c_mod=*/(short)0, c, /*reuse_a=*/false, /*reuse_b=*/false);
}

DEV float red_max16(float v) {
#pragma unroll
  for (int m = 1; m < 16; m <<= 1) v = fmaxf(v, __shfl_xor(v, m, 32));
  return v;
}
DEV float red_sum16(float v) {
#pragma unroll
  for (int m = 1; m < 16; m <<= 1) v += __shfl_xor(v, m, 32);
  return v;
}

// Async Global -> LDS 128-bit copy (CDNA5 GLOBAL_LOAD_ASYNC_TO_LDS_B128).
// lds_off: wave-relative LDS byte offset; gaddr: 64-bit global byte address.
DEV void async_copy_b128(unsigned lds_off, unsigned long long gaddr) {
  asm volatile("global_load_async_to_lds_b128 %0, %1, off"
               :: "v"(lds_off), "v"(gaddr)
               : "memory");
}
DEV void async_wait0() {
  asm volatile("s_wait_asynccnt 0x0" ::: "memory");
}
DEV unsigned lds_offset_of(const void* p) {
  // generic pointer into LDS: low 32 bits == LDS byte offset (aperture rule)
  return (unsigned)(unsigned long long)(uintptr_t)p;
}

// ---------------------------------------------------------------------------
// Projection GEMM: Out[h][nAgent][t][d] (bf16) = alpha * (X[8192x512] @ W[512x512])
// ---------------------------------------------------------------------------
__global__ __launch_bounds__(128) void proj_kernel(
    const float* __restrict__ Xf, const float* __restrict__ Wf,
    unsigned short* __restrict__ Out, float alpha) {
  __shared__ unsigned short sA[64 * 32];  // [row][k]
  __shared__ unsigned short sW[64 * 32];  // [col][k] (transposed)

  const int tid = threadIdx.x, w = tid >> 5, lane = tid & 31;
  const int half = lane >> 4, lq = lane & 15;
  const int r0 = blockIdx.x * 64, n0 = blockIdx.y * 64;

  v8f acc[4];
#pragma unroll
  for (int t = 0; t < 4; ++t) acc[t] = vzero();

  for (int k0 = 0; k0 < 512; k0 += 32) {
    __syncthreads();
#pragma unroll 4
    for (int e = tid; e < 2048; e += 128) {
      int rr = e >> 5, kk = e & 31;
      sA[e] = f2bf(Xf[(size_t)(r0 + rr) * 512 + k0 + kk]);
    }
#pragma unroll 4
    for (int e = tid; e < 2048; e += 128) {
      int kk = e >> 6, cc = e & 63;
      sW[cc * 32 + kk] = f2bf(Wf[(size_t)(k0 + kk) * 512 + n0 + cc]);
    }
    __syncthreads();

    FragBF a;
    frag_load_a(a, (const unsigned int*)sA + (w * 16 + lq) * 16, half);
#pragma unroll
    for (int t = 0; t < 4; ++t) {
      FragBF b;
      frag_load_b(b, (const unsigned int*)sW + (t * 16 + lq) * 16, half);
      acc[t] = wmma_bf16(a, b, acc[t]);
    }
  }

  const int h = n0 >> 6;
#pragma unroll
  for (int t = 0; t < 4; ++t) {
#pragma unroll
    for (int r = 0; r < 8; ++r) {
      const int row = r0 + w * 16 + r + 8 * half;  // r = t_idx*8 + nAgent
      const int nAg = row & 7, tt = row >> 3, d = t * 16 + lq;
      Out[(((size_t)h * 8 + nAg) * 1024 + tt) * 64 + d] = f2bf(acc[t][r] * alpha);
    }
  }
}

// ---------------------------------------------------------------------------
// Agent-aware flash attention.  Block = (lt, n, h), 128 thr (4 waves),
// 64 q-rows per block (16 per wave), S streamed in 64-col tiles.
// ---------------------------------------------------------------------------
__global__ __launch_bounds__(128) void attn_kernel(
    const unsigned short* __restrict__ QS, const unsigned short* __restrict__ QO,
    const unsigned short* __restrict__ KS, const unsigned short* __restrict__ KO,
    const unsigned short* __restrict__ VH,
    const int* __restrict__ qidG, const int* __restrict__ kidG,
    const float* __restrict__ maskG,
    unsigned short* __restrict__ X, float* __restrict__ attnAvg) {
  constexpr int L = 1024, S = 1024, NA = 8;
  __shared__ unsigned short sKs[64 * 64];
  __shared__ unsigned short sKo[64 * 64];
  __shared__ unsigned short sVt[64 * 64];  // transposed: [dv][s]
  __shared__ unsigned short sP[64 * 64];   // probability staging (bf16)
  __shared__ float sMask[64 * 64];
  __shared__ int sQid[64];
  __shared__ int sKid[64];

  const int lt = blockIdx.x, n = blockIdx.y, h = blockIdx.z;
  const int tid = threadIdx.x, w = tid >> 5, lane = tid & 31;
  const int half = lane >> 4, lq = lane & 15;
  const int l0 = lt * 64;
  const size_t hb = ((size_t)(h * NA + n)) * L * 64;  // head-split base (L==S)

  const unsigned ldsKs   = lds_offset_of(&sKs[0]);
  const unsigned ldsKo   = lds_offset_of(&sKo[0]);
  const unsigned ldsMask = lds_offset_of(&sMask[0]);

  // Preload Q fragments (rows for this wave), both self & other, both K-chunks.
  const unsigned int* qsRow =
      (const unsigned int*)(QS + hb) + (size_t)(l0 + w * 16 + lq) * 32;
  const unsigned int* qoRow =
      (const unsigned int*)(QO + hb) + (size_t)(l0 + w * 16 + lq) * 32;
  FragBF fqs0, fqs1, fqo0, fqo1;
  frag_load_a(fqs0, qsRow, half);
  frag_load_a(fqs1, qsRow + 16, half);
  frag_load_a(fqo0, qoRow, half);
  frag_load_a(fqo1, qoRow + 16, half);

  if (tid < 64) sQid[tid] = qidG[(l0 + tid) * NA + n];

  float Mrow[8], Lrow[8];
  v8f Ov[4];
#pragma unroll
  for (int r = 0; r < 8; ++r) { Mrow[r] = -3.0e38f; Lrow[r] = 0.f; }
#pragma unroll
  for (int t = 0; t < 4; ++t) Ov[t] = vzero();

  // ------------------- pass 1: flash softmax + P@V -------------------------
  for (int st = 0; st < 16; ++st) {
    const int s0 = st * 64;
    __syncthreads();
    {
      // K_self / K_other tiles: contiguous 8 KB each -> async b128 copies.
      const unsigned long long gKs =
          (unsigned long long)(uintptr_t)((const char*)(KS + hb) + (size_t)s0 * 128);
      const unsigned long long gKo =
          (unsigned long long)(uintptr_t)((const char*)(KO + hb) + (size_t)s0 * 128);
#pragma unroll
      for (int e = tid; e < 512; e += 128) {
        async_copy_b128(ldsKs + e * 16, gKs + (unsigned long long)e * 16);
        async_copy_b128(ldsKo + e * 16, gKo + (unsigned long long)e * 16);
      }
      // mask tile: 64 rows x 256 B -> async b128 copies (row-strided).
#pragma unroll
      for (int e = tid; e < 1024; e += 128) {
        const int rr = e >> 4, cc = e & 15;
        const unsigned long long gM = (unsigned long long)(uintptr_t)(
            maskG + (size_t)(l0 + rr) * S + s0 + cc * 4);
        async_copy_b128(ldsMask + e * 16, gM);
      }
      // V tile transposed on store (sync path; async copies cannot transpose).
      const unsigned short* gv = VH + hb + (size_t)s0 * 64;
#pragma unroll 4
      for (int e = tid; e < 4096; e += 128) {
        int s = e >> 6, dv = e & 63;
        sVt[dv * 64 + s] = gv[e];
      }
      if (tid < 64) sKid[tid] = kidG[(s0 + tid) * NA + n];
      async_wait0();  // ASYNCcnt -> 0 before signaling the barrier
    }
    __syncthreads();

    v8f cs[4], co[4];
#pragma unroll
    for (int t = 0; t < 4; ++t) { cs[t] = vzero(); co[t] = vzero(); }
#pragma unroll
    for (int t = 0; t < 4; ++t) {
      FragBF b0, b1;
      const unsigned int* kr = (const unsigned int*)sKs + (t * 16 + lq) * 32;
      frag_load_b(b0, kr, half);
      frag_load_b(b1, kr + 16, half);
      cs[t] = wmma_bf16(fqs0, b0, cs[t]);
      cs[t] = wmma_bf16(fqs1, b1, cs[t]);
      const unsigned int* kr2 = (const unsigned int*)sKo + (t * 16 + lq) * 32;
      frag_load_b(b0, kr2, half);
      frag_load_b(b1, kr2 + 16, half);
      co[t] = wmma_bf16(fqo0, b0, co[t]);
      co[t] = wmma_bf16(fqo1, b1, co[t]);
    }
    // agent-aware select + mask
#pragma unroll
    for (int t = 0; t < 4; ++t) {
      const int sl = t * 16 + lq;
      const int kid = sKid[sl];
#pragma unroll
      for (int r = 0; r < 8; ++r) {
        const int m = r + 8 * half;
        float v = (sQid[w * 16 + m] == kid) ? cs[t][r] : co[t][r];
        v += sMask[(w * 16 + m) * 64 + sl];
        cs[t][r] = v;
      }
    }
    // online softmax update
#pragma unroll
    for (int r = 0; r < 8; ++r) {
      float mx = fmaxf(fmaxf(cs[0][r], cs[1][r]), fmaxf(cs[2][r], cs[3][r]));
      mx = red_max16(mx);
      float mnew = fmaxf(Mrow[r], mx);
      float fr = __expf(Mrow[r] - mnew);
      Mrow[r] = mnew;
      float ps = 0.f;
#pragma unroll
      for (int t = 0; t < 4; ++t) {
        float p = __expf(cs[t][r] - mnew);
        cs[t][r] = p;
        ps += p;
      }
      ps = red_sum16(ps);
      Lrow[r] = Lrow[r] * fr + ps;
#pragma unroll
      for (int t = 0; t < 4; ++t) Ov[t][r] *= fr;
    }
    // stage P (C layout -> row-major bf16) in per-wave LDS region
#pragma unroll
    for (int t = 0; t < 4; ++t) {
#pragma unroll
      for (int r = 0; r < 8; ++r) {
        sP[(w * 16 + r + 8 * half) * 64 + t * 16 + lq] = f2bf(cs[t][r]);
      }
    }
    // P @ V   (same-wave LDS: in-order per ISA)
#pragma unroll
    for (int c = 0; c < 2; ++c) {
      FragBF pf;
      frag_load_a(pf, (const unsigned int*)sP + (w * 16 + lq) * 32 + c * 16, half);
#pragma unroll
      for (int t2 = 0; t2 < 4; ++t2) {
        FragBF vf;
        frag_load_b(vf, (const unsigned int*)sVt + (t2 * 16 + lq) * 32 + c * 16, half);
        Ov[t2] = wmma_bf16(pf, vf, Ov[t2]);
      }
    }
  }

  // finalize O, write X (bf16, row-major [l*8+n][512])
  float inv[8];
#pragma unroll
  for (int r = 0; r < 8; ++r) inv[r] = 1.0f / Lrow[r];
#pragma unroll
  for (int t2 = 0; t2 < 4; ++t2) {
#pragma unroll
    for (int r = 0; r < 8; ++r) {
      const int l = l0 + w * 16 + r + 8 * half;
      X[((size_t)l * NA + n) * 512 + h * 64 + t2 * 16 + lq] =
          f2bf(Ov[t2][r] * inv[r]);
    }
  }

  // ------------------- pass 2: head-averaged attention ----------------------
  for (int st = 0; st < 16; ++st) {
    const int s0 = st * 64;
    __syncthreads();
    {
      const unsigned long long gKs =
          (unsigned long long)(uintptr_t)((const char*)(KS + hb) + (size_t)s0 * 128);
      const unsigned long long gKo =
          (unsigned long long)(uintptr_t)((const char*)(KO + hb) + (size_t)s0 * 128);
#pragma unroll
      for (int e = tid; e < 512; e += 128) {
        async_copy_b128(ldsKs + e * 16, gKs + (unsigned long long)e * 16);
        async_copy_b128(ldsKo + e * 16, gKo + (unsigned long long)e * 16);
      }
#pragma unroll
      for (int e = tid; e < 1024; e += 128) {
        const int rr = e >> 4, cc = e & 15;
        const unsigned long long gM = (unsigned long long)(uintptr_t)(
            maskG + (size_t)(l0 + rr) * S + s0 + cc * 4);
        async_copy_b128(ldsMask + e * 16, gM);
      }
      if (tid < 64) sKid[tid] = kidG[(s0 + tid) * NA + n];
      async_wait0();
    }
    __syncthreads();

    v8f cs[4], co[4];
#pragma unroll
    for (int t = 0; t < 4; ++t) { cs[t] = vzero(); co[t] = vzero(); }
#pragma unroll
    for (int t = 0; t < 4; ++t) {
      FragBF b0, b1;
      const unsigned int* kr = (const unsigned int*)sKs + (t * 16 + lq) * 32;
      frag_load_b(b0, kr, half);
      frag_load_b(b1, kr + 16, half);
      cs[t] = wmma_bf16(fqs0, b0, cs[t]);
      cs[t] = wmma_bf16(fqs1, b1, cs[t]);
      const unsigned int* kr2 = (const unsigned int*)sKo + (t * 16 + lq) * 32;
      frag_load_b(b0, kr2, half);
      frag_load_b(b1, kr2 + 16, half);
      co[t] = wmma_bf16(fqo0, b0, co[t]);
      co[t] = wmma_bf16(fqo1, b1, co[t]);
    }
#pragma unroll
    for (int t = 0; t < 4; ++t) {
      const int sl = t * 16 + lq;
      const int kid = sKid[sl];
#pragma unroll
      for (int r = 0; r < 8; ++r) {
        const int m = r + 8 * half;
        float v = (sQid[w * 16 + m] == kid) ? cs[t][r] : co[t][r];
        v += sMask[(w * 16 + m) * 64 + sl];
        const float p = __expf(v - Mrow[r]) * inv[r] * 0.125f;  // /H
        const int l = l0 + w * 16 + m;
        atomicAdd(&attnAvg[((size_t)n * L + l) * S + s0 + sl], p);
      }
    }
  }
}

// ---------------------------------------------------------------------------
// Final FC: out[8192x512] (fp32) = X(bf16) @ Wfc + bfc
// ---------------------------------------------------------------------------
__global__ __launch_bounds__(128) void fc_kernel(
    const unsigned short* __restrict__ Xb, const float* __restrict__ Wf,
    const float* __restrict__ bias, float* __restrict__ Outf) {
  __shared__ unsigned short sA[64 * 32];
  __shared__ unsigned short sW[64 * 32];

  const int tid = threadIdx.x, w = tid >> 5, lane = tid & 31;
  const int half = lane >> 4, lq = lane & 15;
  const int r0 = blockIdx.x * 64, n0 = blockIdx.y * 64;

  v8f acc[4];
#pragma unroll
  for (int t = 0; t < 4; ++t) acc[t] = vzero();

  for (int k0 = 0; k0 < 512; k0 += 32) {
    __syncthreads();
#pragma unroll 4
    for (int e = tid; e < 2048; e += 128) {
      int rr = e >> 5, kk = e & 31;
      sA[e] = Xb[(size_t)(r0 + rr) * 512 + k0 + kk];
    }
#pragma unroll 4
    for (int e = tid; e < 2048; e += 128) {
      int kk = e >> 6, cc = e & 63;
      sW[cc * 32 + kk] = f2bf(Wf[(size_t)(k0 + kk) * 512 + n0 + cc]);
    }
    __syncthreads();

    FragBF a;
    frag_load_a(a, (const unsigned int*)sA + (w * 16 + lq) * 16, half);
#pragma unroll
    for (int t = 0; t < 4; ++t) {
      FragBF b;
      frag_load_b(b, (const unsigned int*)sW + (t * 16 + lq) * 16, half);
      acc[t] = wmma_bf16(a, b, acc[t]);
    }
  }

#pragma unroll
  for (int t = 0; t < 4; ++t) {
#pragma unroll
    for (int r = 0; r < 8; ++r) {
      const int row = r0 + w * 16 + r + 8 * half;
      const int c = n0 + t * 16 + lq;
      Outf[(size_t)row * 512 + c] = acc[t][r] + bias[c];
    }
  }
}

// ---------------------------------------------------------------------------
extern "C" void kernel_launch(void* const* d_in, const int* in_sizes, int n_in,
                              void* d_out, int out_size, void* d_ws, size_t ws_size,
                              hipStream_t stream) {
  (void)in_sizes; (void)n_in; (void)out_size; (void)ws_size;

  const float* q        = (const float*)d_in[0];
  const float* k        = (const float*)d_in[1];
  const float* v        = (const float*)d_in[2];
  const int*   qid      = (const int*)d_in[3];
  const int*   kid      = (const int*)d_in[4];
  const float* mask     = (const float*)d_in[5];
  const float* Wq_self  = (const float*)d_in[6];
  const float* Wq_other = (const float*)d_in[7];
  const float* Wk_self  = (const float*)d_in[8];
  const float* Wk_other = (const float*)d_in[9];
  const float* Wv       = (const float*)d_in[10];
  const float* Wfc      = (const float*)d_in[11];
  const float* bfc      = (const float*)d_in[12];

  unsigned short* ws = (unsigned short*)d_ws;
  const size_t HSZ = (size_t)8 * 8 * 1024 * 64;  // 4M bf16 per tensor
  unsigned short* QS = ws;
  unsigned short* QO = QS + HSZ;
  unsigned short* KS = QO + HSZ;
  unsigned short* KO = KS + HSZ;
  unsigned short* VHp = KO + HSZ;
  unsigned short* X  = VHp + HSZ;

  float* out     = (float*)d_out;
  float* attnAvg = out + (size_t)1024 * 8 * 512;  // [N,L,S]

  const dim3 blk(128);
  const dim3 gB(128, 8);             // 8192/64 row tiles x 512/64 col tiles
  const float scale = 0.125f;        // (512/8)^-0.5

  proj_kernel<<<gB, blk, 0, stream>>>(q, Wq_self,  QS,  scale);
  proj_kernel<<<gB, blk, 0, stream>>>(q, Wq_other, QO,  scale);
  proj_kernel<<<gB, blk, 0, stream>>>(k, Wk_self,  KS,  1.0f);
  proj_kernel<<<gB, blk, 0, stream>>>(k, Wk_other, KO,  1.0f);
  proj_kernel<<<gB, blk, 0, stream>>>(v, Wv,       VHp, 1.0f);

  hipMemsetAsync(attnAvg, 0, (size_t)8 * 1024 * 1024 * sizeof(float), stream);

  attn_kernel<<<dim3(16, 8, 8), blk, 0, stream>>>(QS, QO, KS, KO, VHp,
                                                  qid, kid, mask, X, attnAvg);

  fc_kernel<<<gB, blk, 0, stream>>>(X, Wfc, bfc, out);
}
